// HLSTransformer_54228257079824
// MI455X (gfx1250) — matrix operand order repaired and checked by reading, not compiled, not run on hardware
//
#include <hip/hip_runtime.h>
#include <hip/hip_bf16.h>

// ---------------------------------------------------------------------------
// HLS transformer forward on gfx1250 (MI455X): bf16 WMMA + fp32 accumulate,
// flash-style attention (no [B,N,N] score materialization), per-batch LN.
// ---------------------------------------------------------------------------

#define BQ    32
#define NSEQ  1024
#define FIN   256
#define HDIM  64
#define LN_EPS 1e-5f
#define ATT_SCALE 0.03125f   // 1/sqrt(1024)

typedef __attribute__((ext_vector_type(16))) __bf16 v16bf;
typedef __attribute__((ext_vector_type(8)))  float  v8f;

// ---- WMMA wrapper: D = A(16x32 bf16) * B(32x16 bf16) + C(16x16 f32) -------
__device__ __forceinline__ v8f wmma_bf16(v16bf a, v16bf b, v8f c) {
  return __builtin_amdgcn_wmma_f32_16x16x32_bf16(
      /*neg_a=*/false, a, /*neg_b=*/false, b,
      /*c_mod=*/(short)0, c, /*reuse_a=*/false, /*reuse_b=*/false);
}

// A operand (16x32, 16-bit): lane L holds row M=L%16; hlf=L/16.
// VGPRs 0-3 = K [hlf*8, hlf*8+8), VGPRs 4-7 = K [16+hlf*8, 16+hlf*8+8).
__device__ __forceinline__ v16bf load_opA(const __bf16* row, int k0, int hlf) {
  union { uint4 u[2]; v16bf v; } u;
  u.u[0] = *reinterpret_cast<const uint4*>(row + k0 + hlf * 8);
  u.u[1] = *reinterpret_cast<const uint4*>(row + k0 + 16 + hlf * 8);
  return u.v;
}

// B operand (32x16, 16-bit): lane L holds column N=L%16;
// VGPRs 0-3 = K [hlf*8, hlf*8+8), VGPRs 4-7 = K [16+hlf*8, 16+hlf*8+8).
__device__ __forceinline__ v16bf load_opB(const __bf16* colrun, int k0, int hlf) {
  union { uint4 u[2]; v16bf v; } u;
  const __bf16* p = colrun + k0 + hlf * 8;
  u.u[0] = *reinterpret_cast<const uint4*>(p);
  u.u[1] = *reinterpret_cast<const uint4*>(p + 16);
  return u.v;
}

// ---------------------------------------------------------------------------
// Prep kernels
// ---------------------------------------------------------------------------
__global__ void f32_to_bf16_kernel(const float* __restrict__ in,
                                   __bf16* __restrict__ out, int n) {
  int i = blockIdx.x * blockDim.x + threadIdx.x;
  int stride = gridDim.x * blockDim.x;
  for (; i < n; i += stride) out[i] = (__bf16)in[i];
}

// WT[n*K + k] = W[k*N + n]
__global__ void transpose_to_bf16_kernel(const float* __restrict__ W,
                                         __bf16* __restrict__ WT, int K, int N) {
  int i = blockIdx.x * blockDim.x + threadIdx.x;
  int total = K * N;
  int stride = gridDim.x * blockDim.x;
  for (; i < total; i += stride) {
    int k = i / N, n = i - k * N;
    WT[(size_t)n * K + k] = (__bf16)W[i];
  }
}

// ---------------------------------------------------------------------------
// GEMM: out[M,64] = act(A[M,K] @ B[K,64] + bias) (+resid). One wave = 16x64.
// BT is [64,K] (transposed, bf16). Emits f32, optional bf16, optional bf16^T
// laid out [B, HDIM, NSEQ] (M must be BQ*NSEQ for the transposed emit).
// ---------------------------------------------------------------------------
__global__ void gemm_bias_relu_kernel(const __bf16* __restrict__ A,
                                      const __bf16* __restrict__ BT,
                                      const float*  __restrict__ bias,
                                      const float*  __restrict__ resid,
                                      float*  __restrict__ outf,
                                      __bf16* __restrict__ outb,
                                      __bf16* __restrict__ outbT,
                                      int M, int K) {
  int wave = blockIdx.x * (blockDim.x >> 5) + (threadIdx.x >> 5);
  int lane = threadIdx.x & 31;
  int r = lane & 15, hlf = lane >> 4;
  int m0 = wave * 16;
  if (m0 >= M) return;

  const __bf16* arow = A + (size_t)(m0 + r) * K;
  v8f zero = {};
  v8f c0 = zero, c1 = zero, c2 = zero, c3 = zero;

  for (int k0 = 0; k0 < K; k0 += 32) {
    v16bf a = load_opA(arow, k0, hlf);
    c0 = wmma_bf16(a, load_opB(BT + (size_t)(0 * 16 + r) * K, k0, hlf), c0);
    c1 = wmma_bf16(a, load_opB(BT + (size_t)(1 * 16 + r) * K, k0, hlf), c1);
    c2 = wmma_bf16(a, load_opB(BT + (size_t)(2 * 16 + r) * K, k0, hlf), c2);
    c3 = wmma_bf16(a, load_opB(BT + (size_t)(3 * 16 + r) * K, k0, hlf), c3);
  }

  v8f acc[4] = {c0, c1, c2, c3};
  for (int nt = 0; nt < 4; ++nt) {
    int col = nt * 16 + r;
    float bc = bias[col];
    for (int v = 0; v < 8; ++v) {
      int row = m0 + v + 8 * hlf;
      float val = fmaxf(acc[nt][v] + bc, 0.0f);
      size_t idx = (size_t)row * HDIM + col;
      if (resid) val += resid[idx];
      outf[idx] = val;
      if (outb) outb[idx] = (__bf16)val;
      if (outbT) {
        int bidx = row >> 10;          // / NSEQ
        int n    = row & (NSEQ - 1);
        outbT[((size_t)bidx * HDIM + col) * NSEQ + n] = (__bf16)val;
      }
    }
  }
}

// ---------------------------------------------------------------------------
// Flash attention: out[b,i,:] = softmax((q.k^T + mask)*SCALE) @ v, q=k=v=h.
// One wave per (batch, 16-row query tile); 32 keys per step.
// hb  : [B, NSEQ, HDIM] bf16 (rows)      -> A operand for Q, B operand for K^T
// hbT : [B, HDIM, NSEQ] bf16 (transpose) -> B operand for V
// ---------------------------------------------------------------------------
__global__ void attend_kernel(const __bf16* __restrict__ hb,
                              const __bf16* __restrict__ hbT,
                              const float*  __restrict__ mask,
                              float* __restrict__ out) {
  __shared__ alignas(16) __bf16 pbuf[4][16][32];   // per-wave P staging

  int widx = threadIdx.x >> 5;
  int wave = blockIdx.x * 4 + widx;
  int lane = threadIdx.x & 31;
  int r = lane & 15, hlf = lane >> 4;
  int b  = wave >> 6;        // / (NSEQ/16)
  int it = wave & 63;
  int i0 = it * 16;

  const __bf16* qrow = hb + ((size_t)b * NSEQ + i0 + r) * HDIM;
  v16bf qa0 = load_opA(qrow, 0, hlf);
  v16bf qa1 = load_opA(qrow, 32, hlf);

  v8f zero = {};
  v8f o0 = zero, o1 = zero, o2 = zero, o3 = zero;
  float mrow[8], lrow[8];
  for (int v = 0; v < 8; ++v) { mrow[v] = -__builtin_inff(); lrow[v] = 0.f; }

  for (int j0 = 0; j0 < NSEQ; j0 += 32) {
    // ---- S tile 16x32 = Q(16x64) * K^T ----
    const __bf16* k0row = hb + ((size_t)b * NSEQ + j0 + r) * HDIM;
    const __bf16* k1row = hb + ((size_t)b * NSEQ + j0 + 16 + r) * HDIM;
    v8f s0 = zero, s1 = zero;
    s0 = wmma_bf16(qa0, load_opB(k0row, 0, hlf), s0);
    s0 = wmma_bf16(qa1, load_opB(k0row, 32, hlf), s0);
    s1 = wmma_bf16(qa0, load_opB(k1row, 0, hlf), s1);
    s1 = wmma_bf16(qa1, load_opB(k1row, 32, hlf), s1);

    // ---- (s + mask) * SCALE, block row-max ----
    float p0[8], p1[8], bmax[8];
    for (int v = 0; v < 8; ++v) {
      int i = i0 + v + 8 * hlf;
      size_t mb = ((size_t)b * NSEQ + i) * NSEQ + j0 + r;
      float a0 = (s0[v] + mask[mb]) * ATT_SCALE;
      float a1 = (s1[v] + mask[mb + 16]) * ATT_SCALE;
      p0[v] = a0; p1[v] = a1;
      float t = fmaxf(a0, a1);
      t = fmaxf(t, __shfl_xor(t, 1));
      t = fmaxf(t, __shfl_xor(t, 2));
      t = fmaxf(t, __shfl_xor(t, 4));
      t = fmaxf(t, __shfl_xor(t, 8));
      bmax[v] = t;
    }

    // ---- online softmax update ----
    for (int v = 0; v < 8; ++v) {
      float mnew = fmaxf(mrow[v], bmax[v]);
      float alpha = (mrow[v] == -__builtin_inff()) ? 0.f : __expf(mrow[v] - mnew);
      float e0 = __expf(p0[v] - mnew);
      float e1 = __expf(p1[v] - mnew);
      p0[v] = e0; p1[v] = e1;
      float bs = e0 + e1;
      bs += __shfl_xor(bs, 1);
      bs += __shfl_xor(bs, 2);
      bs += __shfl_xor(bs, 4);
      bs += __shfl_xor(bs, 8);
      lrow[v] = lrow[v] * alpha + bs;
      mrow[v] = mnew;
      o0[v] *= alpha; o1[v] *= alpha; o2[v] *= alpha; o3[v] *= alpha;
    }

    // ---- stage P (C layout -> row-major 16x32 bf16) through LDS ----
    __syncthreads();
    for (int v = 0; v < 8; ++v) {
      int row = v + 8 * hlf;
      pbuf[widx][row][r]      = (__bf16)p0[v];
      pbuf[widx][row][16 + r] = (__bf16)p1[v];
    }
    __syncthreads();
    v16bf pa = load_opA(&pbuf[widx][r][0], 0, hlf);

    // ---- O += P(16x32) * V(32x64); V columns from hbT rows (contiguous) ----
    o0 = wmma_bf16(pa, load_opB(hbT + ((size_t)b * HDIM + 0 * 16 + r) * NSEQ, j0, hlf), o0);
    o1 = wmma_bf16(pa, load_opB(hbT + ((size_t)b * HDIM + 1 * 16 + r) * NSEQ, j0, hlf), o1);
    o2 = wmma_bf16(pa, load_opB(hbT + ((size_t)b * HDIM + 2 * 16 + r) * NSEQ, j0, hlf), o2);
    o3 = wmma_bf16(pa, load_opB(hbT + ((size_t)b * HDIM + 3 * 16 + r) * NSEQ, j0, hlf), o3);
  }

  v8f oacc[4] = {o0, o1, o2, o3};
  for (int v = 0; v < 8; ++v) {
    float inv = 1.0f / lrow[v];
    int row = i0 + v + 8 * hlf;
    float* orow = out + ((size_t)b * NSEQ + row) * HDIM;
    for (int nt = 0; nt < 4; ++nt) orow[nt * 16 + r] = oacc[nt][v] * inv;
  }
}

// ---------------------------------------------------------------------------
// LayerNorm over [NSEQ, HDIM] per batch, fused optional residual add.
// t = a (+ resid);  y = (t - mean)*rsqrt(var+eps)*gamma + beta
// Emits f32, optional bf16 and bf16^T ([B, HDIM, NSEQ]).
// ---------------------------------------------------------------------------
__global__ void ln_kernel(const float* __restrict__ a,
                          const float* __restrict__ resid,
                          float* __restrict__ tbuf,
                          const float* __restrict__ gamma,
                          const float* __restrict__ beta,
                          float*  __restrict__ outf,
                          __bf16* __restrict__ outb,
                          __bf16* __restrict__ outbT) {
  const int NH = NSEQ * HDIM;
  int b = blockIdx.x;
  int tid = threadIdx.x;
  const float* ab = a + (size_t)b * NH;
  const float* rb = resid ? resid + (size_t)b * NH : nullptr;
  float* tb = tbuf + (size_t)b * NH;

  float s = 0.f, ss = 0.f;
  for (int i = tid; i < NH; i += blockDim.x) {
    float t = ab[i] + (rb ? rb[i] : 0.f);
    tb[i] = t;
    s += t; ss += t * t;
  }
  __shared__ float red1[256], red2[256];
  red1[tid] = s; red2[tid] = ss;
  __syncthreads();
  for (int st = 128; st > 0; st >>= 1) {
    if (tid < st) { red1[tid] += red1[tid + st]; red2[tid] += red2[tid + st]; }
    __syncthreads();
  }
  float mean = red1[0] * (1.0f / NH);
  float var  = red2[0] * (1.0f / NH) - mean * mean;
  float rinv = rsqrtf(var + LN_EPS);

  float*  ob  = outf + (size_t)b * NH;
  __bf16* obb = outb  ? outb  + (size_t)b * NH : nullptr;
  __bf16* obt = outbT ? outbT + (size_t)b * NH : nullptr;
  for (int i = tid; i < NH; i += blockDim.x) {
    float y = (tb[i] - mean) * rinv * gamma[i] + beta[i];
    ob[i] = y;
    if (obb) obb[i] = (__bf16)y;
    if (obt) {
      int n = i >> 6, h = i & 63;
      obt[(size_t)h * NSEQ + n] = (__bf16)y;
    }
  }
}

// ---------------------------------------------------------------------------
// pooled = sum_n fc[b,n,:];  out[b] = leaky_relu(pooled . Wout + bout)
// ---------------------------------------------------------------------------
__global__ void pool_head_kernel(const float* __restrict__ fc,
                                 const float* __restrict__ Wout,
                                 const float* __restrict__ bout,
                                 float* __restrict__ out) {
  int b = blockIdx.x, h = threadIdx.x;   // 64 threads
  const float* f = fc + (size_t)b * NSEQ * HDIM;
  float s = 0.f;
  for (int n = 0; n < NSEQ; ++n) s += f[n * HDIM + h];
  __shared__ float red[64];
  red[h] = s * Wout[h];
  __syncthreads();
  for (int st = 32; st > 0; st >>= 1) {
    if (h < st) red[h] += red[h + st];
    __syncthreads();
  }
  if (h == 0) {
    float v = red[0] + bout[0];
    out[b] = v > 0.f ? v : 0.01f * v;
  }
}

// ---------------------------------------------------------------------------
// Host orchestration
// ---------------------------------------------------------------------------
extern "C" void kernel_launch(void* const* d_in, const int* in_sizes, int n_in,
                              void* d_out, int out_size, void* d_ws, size_t ws_size,
                              hipStream_t stream) {
  const float* x    = (const float*)d_in[0];
  const float* mask = (const float*)d_in[1];
  const float* We   = (const float*)d_in[2];
  const float* be   = (const float*)d_in[3];
  const float* gamma= (const float*)d_in[4];
  const float* beta = (const float*)d_in[5];
  const float* W0   = (const float*)d_in[6];
  const float* b0   = (const float*)d_in[7];
  const float* W1   = (const float*)d_in[8];
  const float* b1   = (const float*)d_in[9];
  const float* Wout = (const float*)d_in[10];
  const float* bout = (const float*)d_in[11];
  float* out = (float*)d_out;

  const int M  = BQ * NSEQ;            // 32768
  const int NH = NSEQ * HDIM;          // 65536

  char* ws = (char*)d_ws;
  size_t off = 0;
  auto take = [&](size_t bytes) -> char* {
    char* p = ws + off;
    off = (off + bytes + 255) & ~(size_t)255;
    return p;
  };

  __bf16* xb    = (__bf16*)take((size_t)M * FIN * 2);   // 16 MB
  __bf16* WeT   = (__bf16*)take((size_t)FIN * HDIM * 2);
  __bf16* W0T   = (__bf16*)take((size_t)HDIM * HDIM * 2);
  __bf16* W1T   = (__bf16*)take((size_t)HDIM * HDIM * 2);
  float*  embf  = (float*) take((size_t)BQ * NH * 4);   // 8 MB (kept: block-2 residual)
  __bf16* embb  = (__bf16*)take((size_t)BQ * NH * 2);
  __bf16* embbT = (__bf16*)take((size_t)BQ * NH * 2);
  float*  attn  = (float*) take((size_t)BQ * NH * 4);
  float*  tbuf  = (float*) take((size_t)BQ * NH * 4);
  float*  qkvf  = (float*) take((size_t)BQ * NH * 4);
  float*  fcf   = (float*) take((size_t)BQ * NH * 4);
  __bf16* curb  = (__bf16*)take((size_t)BQ * NH * 2);
  __bf16* curbT = (__bf16*)take((size_t)BQ * NH * 2);

  // --- prep: bf16 copies / transposed weights ---
  f32_to_bf16_kernel<<<4096, 256, 0, stream>>>(x, xb, M * FIN);
  transpose_to_bf16_kernel<<<64, 256, 0, stream>>>(We, WeT, FIN, HDIM);
  transpose_to_bf16_kernel<<<16, 256, 0, stream>>>(W0, W0T, HDIM, HDIM);
  transpose_to_bf16_kernel<<<16, 256, 0, stream>>>(W1, W1T, HDIM, HDIM);

  dim3 gblk(M / 16 / 4), gthr(128);    // 4 waves/block, 1 wave = 16x64 tile
  dim3 ablk(BQ * (NSEQ / 16) / 4);     // 512 blocks

  // --- embed: x_emb = relu(x @ We + be) ---
  gemm_bias_relu_kernel<<<gblk, gthr, 0, stream>>>(
      xb, WeT, be, nullptr, embf, embb, embbT, M, FIN);

  // --- block 1 ---
  attend_kernel<<<ablk, gthr, 0, stream>>>(embb, embbT, mask, attn);
  ln_kernel<<<BQ, 256, 0, stream>>>(attn, embf, tbuf, gamma, beta,
                                    qkvf, curb, curbT);            // qkv1
  gemm_bias_relu_kernel<<<gblk, gthr, 0, stream>>>(
      curb, W0T, b0, qkvf, tbuf, nullptr, nullptr, M, HDIM);       // t = qkv + relu(qkv@W0+b0)
  ln_kernel<<<BQ, 256, 0, stream>>>(tbuf, nullptr, tbuf, gamma, beta,
                                    fcf, curb, curbT);             // fc1

  // --- block 2 (residual is x_emb again) ---
  attend_kernel<<<ablk, gthr, 0, stream>>>(curb, curbT, mask, attn);
  ln_kernel<<<BQ, 256, 0, stream>>>(attn, embf, tbuf, gamma, beta,
                                    qkvf, curb, curbT);            // qkv2
  gemm_bias_relu_kernel<<<gblk, gthr, 0, stream>>>(
      curb, W1T, b1, qkvf, tbuf, nullptr, nullptr, M, HDIM);
  ln_kernel<<<BQ, 256, 0, stream>>>(tbuf, nullptr, tbuf, gamma, beta,
                                    fcf, curb, curbT);             // fc2

  // --- pool + head ---
  pool_head_kernel<<<BQ, 64, 0, stream>>>(fcf, Wout, bout, out);
}
